// DEQDotProductAttentionTransformerMD17_83038897701552
// MI455X (gfx1250) — compile-verified, matrix operand order checked by reading, not run on hardware
//
#include <hip/hip_runtime.h>
#include <hip/hip_bf16.h>
#include <math.h>
#include <stdint.h>

typedef __attribute__((ext_vector_type(16))) _Float16 v16h;
typedef __attribute__((ext_vector_type(8)))  _Float16 v8h;
typedef __attribute__((ext_vector_type(8)))  float    v8f;

static constexpr int NN    = 2048;    // nodes
static constexpr int NE    = 32768;   // edges
static constexpr int NG    = 64;      // graphs
static constexpr int DINJ  = 240;
static constexpr int DEMB  = 480;
static constexpr int DFEAT = 512;
static constexpr int H     = 4;
static constexpr int DH    = 120;
static constexpr int HD    = 480;     // H*DH
static constexpr int HDP   = 512;     // padded stride for q/k/v/agg/z/u/feat
static constexpr int NRBF  = 128;
static constexpr float CUTOFF = 5.0f;

// ---------------------------------------------------------------- utilities
__global__ void k_fill(float* p, float v, int n) {
  int i = blockIdx.x * blockDim.x + threadIdx.x;
  if (i < n) p[i] = v;
}
// padded f32 -> f16 : out[M][Kp], zeros in c >= D
__global__ void k_f32_to_f16_pad(const float* __restrict__ in, _Float16* __restrict__ out,
                                 int M, int D, int ldin, int Kp) {
  int t = blockIdx.x * blockDim.x + threadIdx.x;
  if (t >= M * Kp) return;
  int r = t / Kp, c = t % Kp;
  out[t] = (c < D) ? (_Float16)in[(size_t)r * ldin + c] : (_Float16)0.f;
}
// padded silu(f32) -> f16
__global__ void k_silu_f16_pad(const float* __restrict__ in, _Float16* __restrict__ out,
                               int M, int D, int ldin, int Kp) {
  int t = blockIdx.x * blockDim.x + threadIdx.x;
  if (t >= M * Kp) return;
  int r = t / Kp, c = t % Kp;
  float v = 0.f;
  if (c < D) { float x = in[(size_t)r * ldin + c]; v = x / (1.0f + expf(-x)); }
  out[t] = (_Float16)v;
}
// weight f32 [K][N] -> transposed padded f16 [Np][Kp]
__global__ void k_cvt_wT(const float* __restrict__ src, _Float16* __restrict__ dst,
                         int K, int N, int Kp, int Np) {
  int t = blockIdx.x * blockDim.x + threadIdx.x;
  if (t >= Np * Kp) return;
  int n = t / Kp, k = t % Kp;
  dst[t] = (n < N && k < K) ? (_Float16)src[(size_t)k * N + n] : (_Float16)0.f;
}
// strided add: y[r*ld+c] = a[r*ld+c] + b[r*ld+c], c < D
__global__ void k_add_s(const float* __restrict__ a, const float* __restrict__ b,
                        float* __restrict__ y, int M, int D, int ld) {
  int t = blockIdx.x * blockDim.x + threadIdx.x;
  if (t >= M * D) return;
  int r = t / D, c = t % D;
  size_t i = (size_t)r * ld + c;
  y[i] = a[i] + b[i];
}
__global__ void k_addinp_s(float* __restrict__ a, const float* __restrict__ b,
                           int M, int D, int ld) {
  int t = blockIdx.x * blockDim.x + threadIdx.x;
  if (t >= M * D) return;
  int r = t / D, c = t % D;
  size_t i = (size_t)r * ld + c;
  a[i] += b[i];
}

__device__ inline void atomicMaxFloat(float* addr, float val) {
  unsigned int* ua = (unsigned int*)addr;
  unsigned int old = *ua;
  while (__uint_as_float(old) < val) {
    unsigned int assumed = old;
    old = atomicCAS(ua, assumed, __float_as_uint(val));
    if (old == assumed) break;
  }
}

// ---------------------------------------------------------------- layer norm -> padded f16
__global__ void k_ln_f16(const float* __restrict__ x, _Float16* __restrict__ out,
                         int M, int D, int ld, int Kp) {
  int r = blockIdx.x * blockDim.x + threadIdx.x;
  if (r >= M) return;
  const float* p = x + (size_t)r * ld;
  float mu = 0.f;
  for (int i = 0; i < D; ++i) mu += p[i];
  mu /= (float)D;
  float var = 0.f;
  for (int i = 0; i < D; ++i) { float t = p[i] - mu; var += t * t; }
  var /= (float)D;
  float inv = rsqrtf(var + 1e-6f);
  _Float16* o = out + (size_t)r * Kp;
  for (int i = 0; i < D; ++i) o[i] = (_Float16)((p[i] - mu) * inv);
  for (int i = D; i < Kp; ++i) o[i] = (_Float16)0.f;
}

// ---------------------------------------------------------------- WMMA GEMM
// C[2048][Np] = A[2048][Kp] @ Bt[Np][Kp]^T   (A,Bt f16; C f32)
// Kp % 32 == 0, Np % 64 == 0, M = 2048. No bounds checks anywhere.
// 256 threads = 8 wave32; macrotile 64(M) x 64(N); wave = 16x32 (2 WMMA tiles).
// Double-buffered LDS tiles fed by the CDNA5 direct global->LDS async path:
// next tile's DMA overlaps current tile's WMMAs; s_wait_asynccnt 0x2 uses
// in-order async completion to prove the older (current) tile has landed.
__global__ __launch_bounds__(256) void k_gemm_wmma(
    const _Float16* __restrict__ A, const _Float16* __restrict__ Bt,
    float* __restrict__ C, int Kp, int Np) {
  __shared__ _Float16 As[2][64][48];   // row stride 96B -> 16B aligned chunks
  __shared__ _Float16 Bs[2][64][48];

  const int tid  = threadIdx.x;
  const int lane = tid & 31;
  const int wave = tid >> 5;
  const int wm   = wave >> 1;       // 0..3  (16-row slice)
  const int wn   = wave & 1;        // 0..1  (32-col slice)
  const int m0   = blockIdx.y * 64;
  const int n0   = blockIdx.x * 64;

  const int rA = tid >> 2;          // 0..63
  const int cA = (tid & 3) * 8;     // 0,8,16,24

  const _Float16* gA = A  + (size_t)(m0 + rA) * Kp + cA;
  const _Float16* gB = Bt + (size_t)(n0 + rA) * Kp + cA;
  const unsigned ldsA[2] = { (unsigned)(uintptr_t)(&As[0][rA][cA]),
                             (unsigned)(uintptr_t)(&As[1][rA][cA]) };
  const unsigned ldsB[2] = { (unsigned)(uintptr_t)(&Bs[0][rA][cA]),
                             (unsigned)(uintptr_t)(&Bs[1][rA][cA]) };

  v8f acc0 = {}, acc1 = {};

  // prologue: stage tile 0 into buffer 0
  asm volatile("global_load_async_to_lds_b128 %0, %1, off"
               :: "v"(ldsA[0]), "v"(gA) : "memory");
  asm volatile("global_load_async_to_lds_b128 %0, %1, off"
               :: "v"(ldsB[0]), "v"(gB) : "memory");

  const int nt = Kp >> 5;           // number of 32-wide K tiles
  for (int it = 0; it < nt; ++it) {
    const int cur = it & 1;
    const int nxt = cur ^ 1;
    // last iteration prefetches a clamped dummy tile so the wait stays 0x2
    const int kn = (it + 1 < nt) ? (it + 1) * 32 : 0;

    __syncthreads();                // all waves done reading buffer 'nxt'
    asm volatile("global_load_async_to_lds_b128 %0, %1, off"
                 :: "v"(ldsA[nxt]), "v"(gA + kn) : "memory");
    asm volatile("global_load_async_to_lds_b128 %0, %1, off"
                 :: "v"(ldsB[nxt]), "v"(gB + kn) : "memory");
    // <=2 outstanding: the two older ops (current tile) have completed
    asm volatile("s_wait_asynccnt 0x2" ::: "memory");
    __syncthreads();                // all waves' current tiles visible

    const int row = lane & 15;
    const int kg  = (lane >> 4) * 8;   // lanes 0-15: K{0..7,16..23}; 16-31: K{8..15,24..31}

    v8h alo = *(const v8h*)&As[cur][wm * 16 + row][kg];
    v8h ahi = *(const v8h*)&As[cur][wm * 16 + row][kg + 16];
    v16h a = __builtin_shufflevector(alo, ahi, 0,1,2,3,4,5,6,7,8,9,10,11,12,13,14,15);

    v8h b0lo = *(const v8h*)&Bs[cur][wn * 32 + row][kg];
    v8h b0hi = *(const v8h*)&Bs[cur][wn * 32 + row][kg + 16];
    v16h b0 = __builtin_shufflevector(b0lo, b0hi, 0,1,2,3,4,5,6,7,8,9,10,11,12,13,14,15);
    acc0 = __builtin_amdgcn_wmma_f32_16x16x32_f16(false, a, false, b0,
                                                  (short)0, acc0, false, false);

    v8h b1lo = *(const v8h*)&Bs[cur][wn * 32 + 16 + row][kg];
    v8h b1hi = *(const v8h*)&Bs[cur][wn * 32 + 16 + row][kg + 16];
    v16h b1 = __builtin_shufflevector(b1lo, b1hi, 0,1,2,3,4,5,6,7,8,9,10,11,12,13,14,15);
    acc1 = __builtin_amdgcn_wmma_f32_16x16x32_f16(false, a, false, b1,
                                                  (short)0, acc1, false, false);
  }

  // C layout: VGPR i -> row i + 8*(lane>=16), col = lane&15
  const int col   = lane & 15;
  const int rbase = (lane >> 4) * 8;
#pragma unroll
  for (int i = 0; i < 8; ++i) {
    size_t gm = (size_t)(m0 + wm * 16 + rbase + i);
    C[gm * Np + n0 + wn * 32 + col]      = acc0[i];
    C[gm * Np + n0 + wn * 32 + 16 + col] = acc1[i];
  }
}

// ---------------------------------------------------------------- edge preprocessing
__global__ void k_edge_geom(const float* __restrict__ pos, const int* __restrict__ esrc,
                            const int* __restrict__ edst, float* __restrict__ sh,
                            float* __restrict__ rbf, int E) {
  int e = blockIdx.x * blockDim.x + threadIdx.x;
  if (e >= E) return;
  int s = esrc[e], d = edst[e];
  float vx = pos[s * 3 + 0] - pos[d * 3 + 0];
  float vy = pos[s * 3 + 1] - pos[d * 3 + 1];
  float vz = pos[s * 3 + 2] - pos[d * 3 + 2];
  float dist = sqrtf(vx * vx + vy * vy + vz * vz + 1e-12f);
  float x = vx / dist, y = vy / dist, z = vz / dist;
  const float s3 = 1.7320508075688772f, s15 = 3.872983346207417f, s5 = 2.23606797749979f;
  float* S = sh + (size_t)e * 9;
  S[0] = 1.0f; S[1] = s3 * x; S[2] = s3 * y; S[3] = s3 * z;
  S[4] = s15 * x * y; S[5] = s15 * y * z;
  S[6] = 0.5f * s5 * (3.0f * z * z - 1.0f);
  S[7] = s15 * x * z; S[8] = 0.5f * s15 * (x * x - y * y);
  const float width = CUTOFF / (float)NRBF;
  float* R = rbf + (size_t)e * NRBF;
  for (int i = 0; i < NRBF; ++i) {
    float c = CUTOFF * (float)i / (float)(NRBF - 1);
    float t = (dist - c) / width;
    R[i] = expf(-0.5f * t * t);
  }
}

__global__ void k_edge_rad(const float* __restrict__ rbf, const float* __restrict__ degWr,
                           float* __restrict__ rad, int E) {
  int t = blockIdx.x * blockDim.x + threadIdx.x;
  if (t >= E * 9) return;
  int e = t / 9, s = t % 9;
  const float* R = rbf + (size_t)e * NRBF;
  float acc = 0.f;
  for (int j = 0; j < NRBF; ++j) acc += R[j] * degWr[j * 9 + s];
  rad[t] = acc / (1.0f + expf(-acc));
}

__global__ void k_deg_scatter(const float* __restrict__ sh, const float* __restrict__ rad,
                              const float* __restrict__ degWsh, const int* __restrict__ edst,
                              float* __restrict__ deg, int E) {
  int t = blockIdx.x * blockDim.x + threadIdx.x;
  if (t >= E * DINJ) return;
  int e = t / DINJ, c = t % DINJ;
  const float* S = sh + (size_t)e * 9;
  const float* R = rad + (size_t)e * 9;
  float acc = 0.f;
  for (int s = 0; s < 9; ++s) acc += S[s] * R[s] * degWsh[s * DINJ + c];
  atomicAdd(&deg[(size_t)edst[e] * DINJ + c], acc);
}

__global__ void k_inj(const float* __restrict__ atom_table, const int* __restrict__ node_atom,
                      const float* __restrict__ deg, float* __restrict__ inj, int n) {
  int t = blockIdx.x * blockDim.x + threadIdx.x;
  if (t >= n) return;
  int nd = t / DINJ, c = t % DINJ;
  inj[t] = atom_table[(size_t)node_atom[nd] * DINJ + c] + deg[t] / 16.0f;
}

// zcat[2048][768]: cols 0..479 = z (stride 512), 480..719 = inj
__global__ void k_concat(const float* __restrict__ z, const float* __restrict__ inj,
                         float* __restrict__ zcat, int n) {
  int t = blockIdx.x * blockDim.x + threadIdx.x;
  if (t >= n) return;
  int nd = t / 720, c = t % 720;
  zcat[(size_t)nd * 768 + c] =
      (c < DEMB) ? z[(size_t)nd * HDP + c] : inj[(size_t)nd * DINJ + (c - DEMB)];
}

// ---------------------------------------------------------------- attention (edge kernels)
__global__ void k_edge_r(const float* __restrict__ rbf, const float* __restrict__ Wr,
                         float* __restrict__ redge, int E) {
  int t = blockIdx.x * blockDim.x + threadIdx.x;
  if (t >= E * H) return;
  int e = t / H, h = t % H;
  const float* R = rbf + (size_t)e * NRBF;
  float acc = 0.f;
  for (int j = 0; j < NRBF; ++j) acc += R[j] * Wr[j * H + h];
  redge[t] = acc / (1.0f + expf(-acc));
}

__global__ void k_edge_logit(const float* __restrict__ qb, const float* __restrict__ kb,
                             const float* __restrict__ redge, const float* __restrict__ sh,
                             const float* __restrict__ Wsh, const int* __restrict__ esrc,
                             const int* __restrict__ edst, float* __restrict__ logit, int E) {
  int t = blockIdx.x * blockDim.x + threadIdx.x;
  if (t >= E * H) return;
  int e = t / H, h = t % H;
  int src = esrc[e], dst = edst[e];
  float r = redge[t];
  const float* S = sh + (size_t)e * 9;
  float acc = 0.f;
  for (int dh = 0; dh < DH; ++dh) {
    int col = h * DH + dh;
    float m = 0.f;
    for (int s = 0; s < 9; ++s) m += S[s] * Wsh[s * HD + col];
    float kk = (kb[(size_t)src * HDP + col] + m) * r;
    acc += qb[(size_t)dst * HDP + col] * kk;
  }
  logit[t] = acc * 0.09128709291752768f;  // 1/sqrt(120)
}

__global__ void k_seg_max(const float* __restrict__ logit, const int* __restrict__ edst,
                          float* __restrict__ mx, int E) {
  int t = blockIdx.x * blockDim.x + threadIdx.x;
  if (t >= E * H) return;
  int e = t / H, h = t % H;
  atomicMaxFloat(&mx[edst[e] * H + h], logit[t]);
}

__global__ void k_fix_mx(float* __restrict__ mx, int n) {
  int t = blockIdx.x * blockDim.x + threadIdx.x;
  if (t < n && mx[t] < -1e30f) mx[t] = 0.f;
}

__global__ void k_edge_exp(const float* __restrict__ logit, const float* __restrict__ mx,
                           const int* __restrict__ edst, float* __restrict__ exbuf,
                           float* __restrict__ den, int E) {
  int t = blockIdx.x * blockDim.x + threadIdx.x;
  if (t >= E * H) return;
  int e = t / H, h = t % H;
  float ex = expf(logit[t] - mx[edst[e] * H + h]);
  exbuf[t] = ex;
  atomicAdd(&den[edst[e] * H + h], ex);
}

__global__ void k_edge_agg(const float* __restrict__ exbuf, const float* __restrict__ den,
                           const float* __restrict__ vb, const float* __restrict__ redge,
                           const float* __restrict__ sh, const float* __restrict__ Wsh,
                           const int* __restrict__ esrc, const int* __restrict__ edst,
                           float* __restrict__ agg, int E) {
  int t = blockIdx.x * blockDim.x + threadIdx.x;
  if (t >= E * H) return;
  int e = t / H, h = t % H;
  int src = esrc[e], dst = edst[e];
  float alpha = exbuf[t] / (den[dst * H + h] + 1e-9f);
  float r = redge[t];
  const float* S = sh + (size_t)e * 9;
  for (int dh = 0; dh < DH; ++dh) {
    int col = h * DH + dh;
    float m = 0.f;
    for (int s = 0; s < 9; ++s) m += S[s] * Wsh[s * HD + col];
    float vv = (vb[(size_t)src * HDP + col] + m) * r;
    atomicAdd(&agg[(size_t)dst * HDP + col], alpha * vv);
  }
}

// ---------------------------------------------------------------- readout
__global__ void k_head(const _Float16* __restrict__ uh, const float* __restrict__ hW2,
                       float* __restrict__ enode, int n) {
  int t = blockIdx.x * blockDim.x + threadIdx.x;
  if (t >= n) return;
  const _Float16* U = uh + (size_t)t * DFEAT;
  float acc = 0.f;
  for (int j = 0; j < DFEAT; ++j) acc += (float)U[j] * hW2[j];
  enode[t] = acc;
}

__global__ void k_graph_sum(const float* __restrict__ enode, const int* __restrict__ batch,
                            float* __restrict__ out, int n) {
  int t = blockIdx.x * blockDim.x + threadIdx.x;
  if (t >= n) return;
  atomicAdd(&out[batch[t]], enode[t] * 0.17677669529663687f);  // 1/sqrt(32)
}

// ---------------------------------------------------------------- host orchestration
static inline int cdiv(int a, int b) { return (a + b - 1) / b; }

struct Scratch {
  float *sh, *rbf, *rad, *deg, *inj, *z, *zcat;
  float *q, *k, *v, *redge, *logit, *exbuf, *mx, *den, *agg, *o, *y, *u, *feat, *enode;
  _Float16 *x_h, *t_h, *agg_h, *u_h;
};

// DZ: logical input dim; DZp: its padded stride. zout has stride HDP(=512).
static void run_block(hipStream_t st, const Scratch& S,
                      const int* esrc, const int* edst,
                      const float* zin, int DZ, int DZp, float* zout, int NOUT, bool res,
                      const _Float16* Wq, const _Float16* Wk, const _Float16* Wv,
                      const _Float16* Wo, const _Float16* F1, const _Float16* F2,
                      const float* Wsh, const float* Wr) {
  const int B = 256;
  // x = ln(zin) -> f16 [2048][DZp]
  k_ln_f16<<<cdiv(NN, B), B, 0, st>>>(zin, S.x_h, NN, DZ, DZp, DZp);
  dim3 gq(HDP / 64, NN / 64);
  k_gemm_wmma<<<gq, 256, 0, st>>>(S.x_h, Wq, S.q, DZp, HDP);
  k_gemm_wmma<<<gq, 256, 0, st>>>(S.x_h, Wk, S.k, DZp, HDP);
  k_gemm_wmma<<<gq, 256, 0, st>>>(S.x_h, Wv, S.v, DZp, HDP);
  // edge attention
  k_edge_r<<<cdiv(NE * H, B), B, 0, st>>>(S.rbf, Wr, S.redge, NE);
  k_edge_logit<<<cdiv(NE * H, B), B, 0, st>>>(S.q, S.k, S.redge, S.sh, Wsh, esrc, edst, S.logit, NE);
  k_fill<<<cdiv(NN * H, B), B, 0, st>>>(S.mx, -3.0e38f, NN * H);
  k_seg_max<<<cdiv(NE * H, B), B, 0, st>>>(S.logit, edst, S.mx, NE);
  k_fix_mx<<<cdiv(NN * H, B), B, 0, st>>>(S.mx, NN * H);
  k_fill<<<cdiv(NN * H, B), B, 0, st>>>(S.den, 0.f, NN * H);
  k_edge_exp<<<cdiv(NE * H, B), B, 0, st>>>(S.logit, S.mx, edst, S.exbuf, S.den, NE);
  k_fill<<<cdiv(NN * HDP, B), B, 0, st>>>(S.agg, 0.f, NN * HDP);
  k_edge_agg<<<cdiv(NE * H, B), B, 0, st>>>(S.exbuf, S.den, S.v, S.redge, S.sh, Wsh,
                                            esrc, edst, S.agg, NE);
  // y = zin + agg @ Wo   (o, y stride DZp)
  k_f32_to_f16_pad<<<cdiv(NN * HDP, B), B, 0, st>>>(S.agg, S.agg_h, NN, HD, HDP, HDP);
  dim3 go(DZp / 64, NN / 64);
  k_gemm_wmma<<<go, 256, 0, st>>>(S.agg_h, Wo, S.o, HDP, DZp);
  k_add_s<<<cdiv(NN * DZ, B), B, 0, st>>>(zin, S.o, S.y, NN, DZ, DZp);
  // FFN
  k_ln_f16<<<cdiv(NN, B), B, 0, st>>>(S.y, S.t_h, NN, DZ, DZp, DZp);
  dim3 g1(HDP / 64, NN / 64);
  k_gemm_wmma<<<g1, 256, 0, st>>>(S.t_h, F1, S.u, DZp, HDP);    // u: [2048][512], logical 480
  k_silu_f16_pad<<<cdiv(NN * HDP, B), B, 0, st>>>(S.u, S.u_h, NN, DEMB, HDP, HDP);
  dim3 g2(HDP / 64, NN / 64);
  k_gemm_wmma<<<g2, 256, 0, st>>>(S.u_h, F2, zout, HDP, HDP);   // zout stride 512
  if (res) k_addinp_s<<<cdiv(NN * NOUT, B), B, 0, st>>>(zout, S.y, NN, NOUT, HDP);
}

extern "C" void kernel_launch(void* const* d_in, const int* in_sizes, int n_in,
                              void* d_out, int out_size, void* d_ws, size_t ws_size,
                              hipStream_t stream) {
  (void)in_sizes; (void)n_in; (void)ws_size; (void)out_size;
  const int B = 256;
  // ----- inputs (setup_inputs dict order)
  const float* pos        = (const float*)d_in[0];
  const int*   node_atom  = (const int*)d_in[1];
  const int*   esrc       = (const int*)d_in[2];
  const int*   edst       = (const int*)d_in[3];
  const int*   batch      = (const int*)d_in[4];
  const float* atom_table = (const float*)d_in[5];
  const float* degWr      = (const float*)d_in[6];
  const float* degWsh     = (const float*)d_in[7];
  const float* b0_Wq  = (const float*)d_in[8];
  const float* b0_Wk  = (const float*)d_in[9];
  const float* b0_Wv  = (const float*)d_in[10];
  const float* b0_Wsh = (const float*)d_in[11];
  const float* b0_Wr  = (const float*)d_in[12];
  const float* b0_Wo  = (const float*)d_in[13];
  const float* b0_F1  = (const float*)d_in[14];
  const float* b0_F2  = (const float*)d_in[15];
  const float* bm_Wq  = (const float*)d_in[16];
  const float* bm_Wk  = (const float*)d_in[17];
  const float* bm_Wv  = (const float*)d_in[18];
  const float* bm_Wsh = (const float*)d_in[19];
  const float* bm_Wr  = (const float*)d_in[20];
  const float* bm_Wo  = (const float*)d_in[21];
  const float* bm_F1  = (const float*)d_in[22];
  const float* bm_F2  = (const float*)d_in[23];
  const float* bf_Wq  = (const float*)d_in[24];
  const float* bf_Wk  = (const float*)d_in[25];
  const float* bf_Wv  = (const float*)d_in[26];
  const float* bf_Wsh = (const float*)d_in[27];
  const float* bf_Wr  = (const float*)d_in[28];
  const float* bf_Wo  = (const float*)d_in[29];
  const float* bf_F1  = (const float*)d_in[30];
  const float* bf_F2  = (const float*)d_in[31];
  const float* hW1    = (const float*)d_in[32];
  const float* hW2    = (const float*)d_in[33];
  float* out = (float*)d_out;

  // ----- workspace bump allocator (256B aligned blocks)
  char*  base = (char*)d_ws;
  size_t off  = 0;
  auto alloc = [&](size_t bytes) -> void* {
    off = (off + 255) & ~(size_t)255;
    void* p = base + off;
    off += bytes;
    return p;
  };
  auto allocF = [&](size_t n) { return (float*)alloc(n * 4); };
  auto allocH = [&](size_t n) { return (_Float16*)alloc(n * 2); };

  // transposed padded f16 weights: Wt[Np][Kp]
  auto cvtT = [&](const float* src, int K, int N, int Kp, int Np) {
    _Float16* dst = allocH((size_t)Np * Kp);
    k_cvt_wT<<<cdiv(Np * Kp, B), B, 0, stream>>>(src, dst, K, N, Kp, Np);
    return (const _Float16*)dst;
  };
  // b0: K in {720->768, 480->512}, N in {480->512, 720->768}
  const _Float16* b0Wq_t = cvtT(b0_Wq, 720, 480, 768, 512);
  const _Float16* b0Wk_t = cvtT(b0_Wk, 720, 480, 768, 512);
  const _Float16* b0Wv_t = cvtT(b0_Wv, 720, 480, 768, 512);
  const _Float16* b0Wo_t = cvtT(b0_Wo, 480, 720, 512, 768);
  const _Float16* b0F1_t = cvtT(b0_F1, 720, 480, 768, 512);
  const _Float16* b0F2_t = cvtT(b0_F2, 480, 480, 512, 512);
  const _Float16 *bmWq_t[4], *bmWk_t[4], *bmWv_t[4], *bmWo_t[4], *bmF1_t[4], *bmF2_t[4];
  for (int i = 0; i < 4; ++i) {
    size_t w = (size_t)i * 480 * 480;
    bmWq_t[i] = cvtT(bm_Wq + w, 480, 480, 512, 512);
    bmWk_t[i] = cvtT(bm_Wk + w, 480, 480, 512, 512);
    bmWv_t[i] = cvtT(bm_Wv + w, 480, 480, 512, 512);
    bmWo_t[i] = cvtT(bm_Wo + w, 480, 480, 512, 512);
    bmF1_t[i] = cvtT(bm_F1 + w, 480, 480, 512, 512);
    bmF2_t[i] = cvtT(bm_F2 + w, 480, 480, 512, 512);
  }
  const _Float16* bfWq_t = cvtT(bf_Wq, 480, 480, 512, 512);
  const _Float16* bfWk_t = cvtT(bf_Wk, 480, 480, 512, 512);
  const _Float16* bfWv_t = cvtT(bf_Wv, 480, 480, 512, 512);
  const _Float16* bfWo_t = cvtT(bf_Wo, 480, 480, 512, 512);
  const _Float16* bfF1_t = cvtT(bf_F1, 480, 480, 512, 512);
  const _Float16* bfF2_t = cvtT(bf_F2, 480, 512, 512, 512);
  const _Float16* hW1_t  = cvtT(hW1, 512, 512, 512, 512);

  // activation scratch
  Scratch S;
  S.sh    = allocF((size_t)NE * 9);
  S.rbf   = allocF((size_t)NE * NRBF);
  S.rad   = allocF((size_t)NE * 9);
  S.deg   = allocF((size_t)NN * DINJ);
  S.inj   = allocF((size_t)NN * DINJ);
  S.z     = allocF((size_t)NN * HDP);      // [2048][512], logical 480
  S.zcat  = allocF((size_t)NN * 768);      // [2048][768], logical 720
  S.q     = allocF((size_t)NN * HDP);
  S.k     = allocF((size_t)NN * HDP);
  S.v     = allocF((size_t)NN * HDP);
  S.redge = allocF((size_t)NE * H);
  S.logit = allocF((size_t)NE * H);
  S.exbuf = allocF((size_t)NE * H);
  S.mx    = allocF((size_t)NN * H);
  S.den   = allocF((size_t)NN * H);
  S.agg   = allocF((size_t)NN * HDP);
  S.o     = allocF((size_t)NN * 768);
  S.y     = allocF((size_t)NN * 768);
  S.u     = allocF((size_t)NN * HDP);
  S.feat  = allocF((size_t)NN * HDP);      // [2048][512]
  S.enode = allocF((size_t)NN);
  S.x_h   = allocH((size_t)NN * 768);
  S.t_h   = allocH((size_t)NN * 768);
  S.agg_h = allocH((size_t)NN * HDP);
  S.u_h   = allocH((size_t)NN * HDP);

  // ----- preprocessing
  k_edge_geom<<<cdiv(NE, B), B, 0, stream>>>(pos, esrc, edst, S.sh, S.rbf, NE);
  k_edge_rad<<<cdiv(NE * 9, B), B, 0, stream>>>(S.rbf, degWr, S.rad, NE);
  k_fill<<<cdiv(NN * DINJ, B), B, 0, stream>>>(S.deg, 0.f, NN * DINJ);
  k_deg_scatter<<<cdiv(NE * DINJ, B), B, 0, stream>>>(S.sh, S.rad, degWsh, edst, S.deg, NE);
  k_inj<<<cdiv(NN * DINJ, B), B, 0, stream>>>(atom_table, node_atom, S.deg, S.inj, NN * DINJ);
  k_fill<<<cdiv(NN * HDP, B), B, 0, stream>>>(S.z, 0.f, NN * HDP);

  // ----- iterated transformer stack
  for (int it = 0; it < 2; ++it) {
    k_concat<<<cdiv(NN * 720, B), B, 0, stream>>>(S.z, S.inj, S.zcat, NN * 720);
    run_block(stream, S, esrc, edst, S.zcat, 720, 768, S.z, DEMB, /*res=*/false,
              b0Wq_t, b0Wk_t, b0Wv_t, b0Wo_t, b0F1_t, b0F2_t, b0_Wsh, b0_Wr);
    for (int i = 0; i < 4; ++i) {
      run_block(stream, S, esrc, edst, S.z, DEMB, HDP, S.z, DEMB, /*res=*/true,
                bmWq_t[i], bmWk_t[i], bmWv_t[i], bmWo_t[i], bmF1_t[i], bmF2_t[i],
                bm_Wsh + (size_t)i * 9 * 480, bm_Wr + (size_t)i * 128 * 4);
    }
  }
  // final feature block
  run_block(stream, S, esrc, edst, S.z, DEMB, HDP, S.feat, DFEAT, /*res=*/false,
            bfWq_t, bfWk_t, bfWv_t, bfWo_t, bfF1_t, bfF2_t, bf_Wsh, bf_Wr);

  // ----- readout head
  k_ln_f16<<<cdiv(NN, B), B, 0, stream>>>(S.feat, S.t_h, NN, DFEAT, HDP, HDP);
  dim3 gh(HDP / 64, NN / 64);
  k_gemm_wmma<<<gh, 256, 0, stream>>>(S.t_h, hW1_t, S.u, HDP, HDP);
  k_silu_f16_pad<<<cdiv(NN * HDP, B), B, 0, stream>>>(S.u, S.u_h, NN, DFEAT, HDP, HDP);
  k_head<<<cdiv(NN, B), B, 0, stream>>>(S.u_h, hW2, S.enode, NN);
  k_fill<<<1, 64, 0, stream>>>(out, 0.f, NG);
  k_graph_sum<<<cdiv(NN, B), B, 0, stream>>>(S.enode, batch, out, NN);
}
